// S4DPLRTransitionModel_30305289240698
// MI455X (gfx1250) — compile-verified, matrix operand order checked by reading, not run on hardware
//
#include <hip/hip_runtime.h>
#include <hip/hip_bf16.h>

// ---------------- problem dimensions ----------------
#define BQ    2048      // batch
#define DD    512       // latent dim
#define CDim  256       // complex dim
#define RR    8         // low rank
#define UDim  32        // u dim
#define NYq   50        // n_y
#define SELIN 545       // D + UD + 1
#define SELP  576       // SELIN padded to multiple of 32
#define HH    128       // hidden

typedef __attribute__((ext_vector_type(16))) __bf16 v16bf;
typedef __attribute__((ext_vector_type(8)))  float  v8f;

union Frag16 { v16bf v; uint4 u[2]; };
union FAcc   { v8f   v; float f[8]; };

// ---------------- build X = concat(zt, ut, dt) padded to 576, bf16 ----------
__global__ __launch_bounds__(256) void k_build_x(
    const float* __restrict__ zt, const float* __restrict__ dt,
    const float* __restrict__ ut, __hip_bfloat16* __restrict__ X) {
  const int n = BQ * SELP;
  int i = blockIdx.x * blockDim.x + threadIdx.x;
  if (i >= n) return;
  const int b = i / SELP, c = i % SELP;
  float v;
  if (c < DD)              v = zt[(size_t)b * DD + c];
  else if (c < DD + UDim)  v = ut[(size_t)b * UDim + (c - DD)];
  else if (c == DD + UDim) v = dt[b];
  else                     v = 0.f;
  X[i] = __float2bfloat16(v);
}

// ------- W[K,N] fp32 -> Wt[N,Kp] bf16 (transposed, K zero-padded to Kp) -----
__global__ __launch_bounds__(256) void k_tconv(
    const float* __restrict__ W, __hip_bfloat16* __restrict__ Wt,
    int K, int N, int Kp) {
  const long n = (long)N * Kp;
  long i = (long)blockIdx.x * blockDim.x + threadIdx.x;
  if (i >= n) return;
  const int nn = (int)(i / Kp), k = (int)(i % Kp);
  Wt[i] = __float2bfloat16(k < K ? W[(size_t)k * N + nn] : 0.f);
}

// ---------------- fp32 -> bf16 ----------------
__global__ __launch_bounds__(256) void k_cvt_bf16(
    const float* __restrict__ src, __hip_bfloat16* __restrict__ dst, int n) {
  int i = blockIdx.x * blockDim.x + threadIdx.x;
  if (i >= n) return;
  dst[i] = __float2bfloat16(src[i]);
}

// ---------------- WMMA GEMM: C[M,N] = A[M,Kp](bf16) * Bt[N,Kp]^T + bias -----
// One wave computes a 32x64 tile: 2 A row-fragments x 4 B col-fragments
// -> 8 WMMA per K-step sharing 12 b128 loads (B frags reused across rows).
// A-frag layout (ISA 7.12.2, 16-bit A 16x32): lanes 0-15 hold K {0..7,16..23},
// lanes 16-31 hold K {8..15,24..31}; B-frag mirrors with N across lanes.
__global__ __launch_bounds__(256) void k_wmma_gemm_bias(
    const __hip_bfloat16* __restrict__ A, const __hip_bfloat16* __restrict__ Bt,
    const float* __restrict__ bias, float* __restrict__ C,
    int M, int N, int Kp) {
  const int lane  = threadIdx.x & 31;
  const int tile  = blockIdx.x * 8 + (threadIdx.x >> 5);
  const int ncolT = N >> 6;
  if (tile >= (M >> 5) * ncolT) return;       // wave-uniform guard (EXEC stays full)
  const int mt    = tile / ncolT;
  const int nt    = tile % ncolT;
  const int lrow  = lane & 15;
  const int khalf = lane >> 4;
  const int nbase = nt << 6;

  const uint4* Ar0 = (const uint4*)(A  + (size_t)((mt << 5) +  0 + lrow) * Kp);
  const uint4* Ar1 = (const uint4*)(A  + (size_t)((mt << 5) + 16 + lrow) * Kp);
  const uint4* B0  = (const uint4*)(Bt + (size_t)(nbase +  0 + lrow) * Kp);
  const uint4* B1  = (const uint4*)(Bt + (size_t)(nbase + 16 + lrow) * Kp);
  const uint4* B2  = (const uint4*)(Bt + (size_t)(nbase + 32 + lrow) * Kp);
  const uint4* B3  = (const uint4*)(Bt + (size_t)(nbase + 48 + lrow) * Kp);

  FAcc a00{}, a01{}, a02{}, a03{};   // rows 0..15 of tile
  FAcc a10{}, a11{}, a12{}, a13{};   // rows 16..31 of tile
  const int ksteps = Kp >> 5;
  for (int ks = 0; ks < ksteps; ++ks) {
    const int i1 = (ks << 2) + khalf;   // uint4 index of first 8 bf16
    const int i2 = i1 + 2;              // +16 elements
    Frag16 af0, af1, bf0, bf1, bf2, bf3;
    af0.u[0] = Ar0[i1]; af0.u[1] = Ar0[i2];
    af1.u[0] = Ar1[i1]; af1.u[1] = Ar1[i2];
    bf0.u[0] = B0[i1];  bf0.u[1] = B0[i2];
    bf1.u[0] = B1[i1];  bf1.u[1] = B1[i2];
    bf2.u[0] = B2[i1];  bf2.u[1] = B2[i2];
    bf3.u[0] = B3[i1];  bf3.u[1] = B3[i2];
    __builtin_prefetch((const void*)(Ar0 + i1 + 4), 0, 1);   // next K-step
    a00.v = __builtin_amdgcn_wmma_f32_16x16x32_bf16(false, af0.v, false, bf0.v,
                                                    (short)0, a00.v, false, false);
    a01.v = __builtin_amdgcn_wmma_f32_16x16x32_bf16(false, af0.v, false, bf1.v,
                                                    (short)0, a01.v, false, false);
    a02.v = __builtin_amdgcn_wmma_f32_16x16x32_bf16(false, af0.v, false, bf2.v,
                                                    (short)0, a02.v, false, false);
    a03.v = __builtin_amdgcn_wmma_f32_16x16x32_bf16(false, af0.v, false, bf3.v,
                                                    (short)0, a03.v, false, false);
    a10.v = __builtin_amdgcn_wmma_f32_16x16x32_bf16(false, af1.v, false, bf0.v,
                                                    (short)0, a10.v, false, false);
    a11.v = __builtin_amdgcn_wmma_f32_16x16x32_bf16(false, af1.v, false, bf1.v,
                                                    (short)0, a11.v, false, false);
    a12.v = __builtin_amdgcn_wmma_f32_16x16x32_bf16(false, af1.v, false, bf2.v,
                                                    (short)0, a12.v, false, false);
    a13.v = __builtin_amdgcn_wmma_f32_16x16x32_bf16(false, af1.v, false, bf3.v,
                                                    (short)0, a13.v, false, false);
  }
  const float bv0 = bias[nbase +  0 + lrow];
  const float bv1 = bias[nbase + 16 + lrow];
  const float bv2 = bias[nbase + 32 + lrow];
  const float bv3 = bias[nbase + 48 + lrow];
  // C/D layout: lane = N column; VGPR i -> row M = i + 8*khalf
  float* Cr0 = C + (size_t)((mt << 5) +  0 + (khalf << 3)) * N + nbase + lrow;
  float* Cr1 = C + (size_t)((mt << 5) + 16 + (khalf << 3)) * N + nbase + lrow;
  for (int i = 0; i < 8; ++i) {
    float* p0 = Cr0 + (size_t)i * N;
    float* p1 = Cr1 + (size_t)i * N;
    p0[0]  = a00.f[i] + bv0;
    p0[16] = a01.f[i] + bv1;
    p0[32] = a02.f[i] + bv2;
    p0[48] = a03.f[i] + bv3;
    p1[0]  = a10.f[i] + bv0;
    p1[16] = a11.f[i] + bv1;
    p1[32] = a12.f[i] + bv2;
    p1[48] = a13.f[i] + bv3;
  }
}

// ---------------- BatchNorm column stats (biased var, batch mode) -----------
__global__ __launch_bounds__(256) void k_bn_stats(
    const float* __restrict__ Y, float* __restrict__ mean,
    float* __restrict__ rinv, int Mrows, int Ncols) {
  const int j = blockIdx.x;
  float s = 0.f, s2 = 0.f;
  for (int r = threadIdx.x; r < Mrows; r += blockDim.x) {
    float v = Y[(size_t)r * Ncols + j];
    s += v; s2 += v * v;
  }
  __shared__ float ss[256], ss2[256];
  ss[threadIdx.x] = s; ss2[threadIdx.x] = s2;
  __syncthreads();
  for (int o = 128; o; o >>= 1) {
    if (threadIdx.x < o) { ss[threadIdx.x] += ss[threadIdx.x + o];
                           ss2[threadIdx.x] += ss2[threadIdx.x + o]; }
    __syncthreads();
  }
  if (threadIdx.x == 0) {
    float mu  = ss[0] / Mrows;
    float var = ss2[0] / Mrows - mu * mu;
    mean[j] = mu;
    rinv[j] = rsqrtf(var + 1e-5f);
  }
}

// ---------------- BN apply + ReLU -> bf16 ----------------
__global__ __launch_bounds__(256) void k_bn_apply(
    const float* __restrict__ Y, const float* __restrict__ mean,
    const float* __restrict__ rinv, const float* __restrict__ g,
    const float* __restrict__ be, __hip_bfloat16* __restrict__ H,
    int n, int Ncols) {
  int i = blockIdx.x * blockDim.x + threadIdx.x;
  if (i >= n) return;
  const int j = i % Ncols;
  float v = (Y[i] - mean[j]) * rinv[j] * g[j] + be[j];
  H[i] = __float2bfloat16(v > 0.f ? v : 0.f);
}

// ---------------- fused per-row combine ----------------
__global__ __launch_bounds__(256) void k_combine(
    const float* __restrict__ zt, const float* __restrict__ dt,
    const float* __restrict__ ut,
    const float* __restrict__ Pa,  const float* __restrict__ Po,
    const float* __restrict__ PUr, const float* __restrict__ PUi,
    const float* __restrict__ PVr, const float* __restrict__ PVi,
    const float* __restrict__ PBr, const float* __restrict__ PBi,
    const float* __restrict__ PCr, const float* __restrict__ PCi,
    const float* __restrict__ PD,
    float* __restrict__ out_z, float* __restrict__ out_y) {
  const int b   = blockIdx.x;
  const int tid = threadIdx.x;
  __shared__ float s_ut[UDim];
  __shared__ float s_tre[RR], s_tim[RR];
  __shared__ float s_zr[CDim], s_zi[CDim];

  const float dtv = dt[b];
  if (tid < UDim) s_ut[tid] = ut[(size_t)b * UDim + tid] * dtv;
  if (tid < RR)       s_tre[tid] = 0.f;
  else if (tid < 2*RR) s_tim[tid - RR] = 0.f;
  __syncthreads();

  const int d = tid;                       // blockDim == CDim == 256
  const float zr = zt[(size_t)b * DD + d];
  const float zi = zt[(size_t)b * DD + CDim + d];

  // t[r] = sum_d conj(V[d,r]) * z[d]
  {
    const float* vr = PVr + (size_t)b * (CDim * RR) + (size_t)d * RR;
    const float* vi = PVi + (size_t)b * (CDim * RR) + (size_t)d * RR;
    float tre[RR], tim[RR];
    #pragma unroll
    for (int r = 0; r < RR; ++r) {
      tre[r] = vr[r] * zr + vi[r] * zi;    // Re(conj(v) z)
      tim[r] = vr[r] * zi - vi[r] * zr;    // Im(conj(v) z)
    }
    #pragma unroll
    for (int r = 0; r < RR; ++r) {
      #pragma unroll
      for (int o = 16; o; o >>= 1) {
        tre[r] += __shfl_xor(tre[r], o);
        tim[r] += __shfl_xor(tim[r], o);
      }
    }
    if ((tid & 31) == 0) {
      #pragma unroll
      for (int r = 0; r < RR; ++r) {
        atomicAdd(&s_tre[r], tre[r]);
        atomicAdd(&s_tim[r], tim[r]);
      }
    }
  }
  __syncthreads();

  // z_next[d] = Lam*z + U t + B (ut*dt)
  {
    const float av = Pa[(size_t)b * CDim + d];
    const float sp = (av > 20.f) ? av : log1pf(expf(av));   // softplus
    const float mag = expf(-sp);                            // exp(alpha), alpha=-softplus
    const float om  = Po[(size_t)b * CDim + d];
    const float lr = mag * cosf(om);
    const float li = mag * sinf(om);
    float znr = lr * zr - li * zi;
    float zni = lr * zi + li * zr;

    const float* urp = PUr + (size_t)b * (CDim * RR) + (size_t)d * RR;
    const float* uip = PUi + (size_t)b * (CDim * RR) + (size_t)d * RR;
    #pragma unroll
    for (int r = 0; r < RR; ++r) {
      znr += urp[r] * s_tre[r] - uip[r] * s_tim[r];
      zni += urp[r] * s_tim[r] + uip[r] * s_tre[r];
    }
    const float* brp = PBr + (size_t)b * (CDim * UDim) + (size_t)d * UDim;
    const float* bip = PBi + (size_t)b * (CDim * UDim) + (size_t)d * UDim;
    #pragma unroll
    for (int u = 0; u < UDim; ++u) {
      const float uu = s_ut[u];
      znr += brp[u] * uu;
      zni += bip[u] * uu;
    }
    out_z[(size_t)b * DD + d]        = znr;
    out_z[(size_t)b * DD + CDim + d] = zni;
    s_zr[d] = znr; s_zi[d] = zni;
  }
  __syncthreads();

  // yt[n] = Re(C z_next) + D (ut*dt)
  for (int n = tid; n < NYq; n += blockDim.x) {
    const float* cr = PCr + (size_t)b * (NYq * CDim) + (size_t)n * CDim;
    const float* ci = PCi + (size_t)b * (NYq * CDim) + (size_t)n * CDim;
    float acc = 0.f;
    for (int k = 0; k < CDim; ++k) acc += cr[k] * s_zr[k] - ci[k] * s_zi[k];
    const float* dp = PD + (size_t)b * (NYq * UDim) + (size_t)n * UDim;
    #pragma unroll
    for (int u = 0; u < UDim; ++u) acc += dp[u] * s_ut[u];
    out_y[(size_t)b * NYq + n] = acc;
  }
}

// ============================================================================
extern "C" void kernel_launch(void* const* d_in, const int* in_sizes, int n_in,
                              void* d_out, int out_size, void* d_ws, size_t ws_size,
                              hipStream_t stream) {
  (void)in_sizes; (void)n_in; (void)out_size; (void)ws_size;
  const float* zt  = (const float*)d_in[0];
  const float* dtp = (const float*)d_in[1];
  const float* ut  = (const float*)d_in[2];
  const float* W0  = (const float*)d_in[3];  const float* b0  = (const float*)d_in[4];
  const float* W1  = (const float*)d_in[5];  const float* b1  = (const float*)d_in[6];
  const float* W2  = (const float*)d_in[7];  const float* b2  = (const float*)d_in[8];
  const float* Wa  = (const float*)d_in[9];  const float* ba  = (const float*)d_in[10];
  const float* Wo  = (const float*)d_in[11]; const float* bo  = (const float*)d_in[12];
  const float* WUr = (const float*)d_in[13]; const float* bUr = (const float*)d_in[14];
  const float* WUi = (const float*)d_in[15]; const float* bUi = (const float*)d_in[16];
  const float* WVr = (const float*)d_in[17]; const float* bVr = (const float*)d_in[18];
  const float* WVi = (const float*)d_in[19]; const float* bVi = (const float*)d_in[20];
  const float* WBr = (const float*)d_in[21]; const float* bBr = (const float*)d_in[22];
  const float* WBi = (const float*)d_in[23]; const float* bBi = (const float*)d_in[24];
  const float* WCr = (const float*)d_in[25]; const float* bCr = (const float*)d_in[26];
  const float* WCi = (const float*)d_in[27]; const float* bCi = (const float*)d_in[28];
  const float* WD  = (const float*)d_in[29]; const float* bD  = (const float*)d_in[30];
  const float* g0  = (const float*)d_in[31]; const float* be0 = (const float*)d_in[32];
  const float* g1  = (const float*)d_in[33]; const float* be1 = (const float*)d_in[34];

  char* ws = (char*)d_ws;
  size_t off = 0;
  auto alloc = [&](size_t bytes) -> void* {
    void* p = ws + off;
    off = (off + bytes + 255) & ~(size_t)255;
    return p;
  };
  typedef __hip_bfloat16 bf;

  // bf16 staging buffers
  bf* Xbf  = (bf*)alloc((size_t)BQ * SELP * 2);
  bf* W0t  = (bf*)alloc((size_t)HH * SELP * 2);
  bf* W1t  = (bf*)alloc((size_t)HH * HH * 2);
  bf* W2t  = (bf*)alloc((size_t)DD * HH * 2);
  bf* Wat  = (bf*)alloc((size_t)CDim * DD * 2);
  bf* Wot  = (bf*)alloc((size_t)CDim * DD * 2);
  bf* WUrt = (bf*)alloc((size_t)CDim * RR * DD * 2);
  bf* WUit = (bf*)alloc((size_t)CDim * RR * DD * 2);
  bf* WVrt = (bf*)alloc((size_t)CDim * RR * DD * 2);
  bf* WVit = (bf*)alloc((size_t)CDim * RR * DD * 2);
  bf* WBrt = (bf*)alloc((size_t)CDim * UDim * DD * 2);
  bf* WBit = (bf*)alloc((size_t)CDim * UDim * DD * 2);
  bf* WCrt = (bf*)alloc((size_t)NYq * CDim * DD * 2);
  bf* WCit = (bf*)alloc((size_t)NYq * CDim * DD * 2);
  bf* WDt  = (bf*)alloc((size_t)NYq * UDim * DD * 2);
  // activations
  float* y0    = (float*)alloc((size_t)BQ * HH * 4);
  float* y1    = (float*)alloc((size_t)BQ * HH * 4);
  bf*    h0bf  = (bf*)alloc((size_t)BQ * HH * 2);
  bf*    h1bf  = (bf*)alloc((size_t)BQ * HH * 2);
  float* hz    = (float*)alloc((size_t)BQ * DD * 4);
  bf*    hzbf  = (bf*)alloc((size_t)BQ * DD * 2);
  float* mean0 = (float*)alloc(HH * 4); float* inv0 = (float*)alloc(HH * 4);
  float* mean1 = (float*)alloc(HH * 4); float* inv1 = (float*)alloc(HH * 4);
  // projections (fp32)
  float* Pa  = (float*)alloc((size_t)BQ * CDim * 4);
  float* Po  = (float*)alloc((size_t)BQ * CDim * 4);
  float* PUr = (float*)alloc((size_t)BQ * CDim * RR * 4);
  float* PUi = (float*)alloc((size_t)BQ * CDim * RR * 4);
  float* PVr = (float*)alloc((size_t)BQ * CDim * RR * 4);
  float* PVi = (float*)alloc((size_t)BQ * CDim * RR * 4);
  float* PBr = (float*)alloc((size_t)BQ * CDim * UDim * 4);
  float* PBi = (float*)alloc((size_t)BQ * CDim * UDim * 4);
  float* PCr = (float*)alloc((size_t)BQ * NYq * CDim * 4);
  float* PCi = (float*)alloc((size_t)BQ * NYq * CDim * 4);
  float* PD  = (float*)alloc((size_t)BQ * NYq * UDim * 4);

  auto tconv = [&](const float* W, bf* Wt, int K, int N, int Kp) {
    long n = (long)N * Kp;
    k_tconv<<<dim3((unsigned)((n + 255) / 256)), dim3(256), 0, stream>>>(W, Wt, K, N, Kp);
  };
  auto gemm = [&](const bf* A, const bf* Bt, const float* bias, float* C,
                  int M, int N, int Kp) {
    int tiles = (M / 32) * (N / 64);
    k_wmma_gemm_bias<<<dim3((tiles + 7) / 8), dim3(256), 0, stream>>>(
        A, Bt, bias, C, M, N, Kp);
  };

  // stage inputs + weights as bf16
  k_build_x<<<dim3((BQ * SELP + 255) / 256), dim3(256), 0, stream>>>(zt, dtp, ut, Xbf);
  tconv(W0,  W0t,  SELIN, HH,            SELP);
  tconv(W1,  W1t,  HH,    HH,            HH);
  tconv(W2,  W2t,  HH,    DD,            HH);
  tconv(Wa,  Wat,  DD,    CDim,          DD);
  tconv(Wo,  Wot,  DD,    CDim,          DD);
  tconv(WUr, WUrt, DD,    CDim * RR,     DD);
  tconv(WUi, WUit, DD,    CDim * RR,     DD);
  tconv(WVr, WVrt, DD,    CDim * RR,     DD);
  tconv(WVi, WVit, DD,    CDim * RR,     DD);
  tconv(WBr, WBrt, DD,    CDim * UDim,   DD);
  tconv(WBi, WBit, DD,    CDim * UDim,   DD);
  tconv(WCr, WCrt, DD,    NYq * CDim,    DD);
  tconv(WCi, WCit, DD,    NYq * CDim,    DD);
  tconv(WD,  WDt,  DD,    NYq * UDim,    DD);

  // selector MLP with BatchNorm (batch stats) + ReLU
  gemm(Xbf, W0t, b0, y0, BQ, HH, SELP);
  k_bn_stats<<<dim3(HH), dim3(256), 0, stream>>>(y0, mean0, inv0, BQ, HH);
  k_bn_apply<<<dim3((BQ * HH + 255) / 256), dim3(256), 0, stream>>>(
      y0, mean0, inv0, g0, be0, h0bf, BQ * HH, HH);
  gemm(h0bf, W1t, b1, y1, BQ, HH, HH);
  k_bn_stats<<<dim3(HH), dim3(256), 0, stream>>>(y1, mean1, inv1, BQ, HH);
  k_bn_apply<<<dim3((BQ * HH + 255) / 256), dim3(256), 0, stream>>>(
      y1, mean1, inv1, g1, be1, h1bf, BQ * HH, HH);
  gemm(h1bf, W2t, b2, hz, BQ, DD, HH);
  k_cvt_bf16<<<dim3((BQ * DD + 255) / 256), dim3(256), 0, stream>>>(hz, hzbf, BQ * DD);

  // projection fan (all K = 512, L2-resident weights)
  gemm(hzbf, Wat,  ba,  Pa,  BQ, CDim,        DD);
  gemm(hzbf, Wot,  bo,  Po,  BQ, CDim,        DD);
  gemm(hzbf, WUrt, bUr, PUr, BQ, CDim * RR,   DD);
  gemm(hzbf, WUit, bUi, PUi, BQ, CDim * RR,   DD);
  gemm(hzbf, WVrt, bVr, PVr, BQ, CDim * RR,   DD);
  gemm(hzbf, WVit, bVi, PVi, BQ, CDim * RR,   DD);
  gemm(hzbf, WBrt, bBr, PBr, BQ, CDim * UDim, DD);
  gemm(hzbf, WBit, bBi, PBi, BQ, CDim * UDim, DD);
  gemm(hzbf, WCrt, bCr, PCr, BQ, NYq * CDim,  DD);
  gemm(hzbf, WCit, bCi, PCi, BQ, NYq * CDim,  DD);
  gemm(hzbf, WDt,  bD,  PD,  BQ, NYq * UDim,  DD);

  // fused combine: Lam, low-rank, B-input, outputs
  float* out_z = (float*)d_out;
  float* out_y = (float*)d_out + (size_t)BQ * DD;
  k_combine<<<dim3(BQ), dim3(256), 0, stream>>>(
      zt, dtp, ut, Pa, Po, PUr, PUi, PVr, PVi, PBr, PBi, PCr, PCi, PD,
      out_z, out_y);
}
